// MultiHeadAttention_69166153334828
// MI455X (gfx1250) — matrix operand order repaired and checked by reading, not compiled, not run on hardware
//
#include <hip/hip_runtime.h>
#include <hip/hip_bf16.h>

// ---------------------------------------------------------------------------
// MI455X (gfx1250) causal multi-head attention, bf16 WMMA everywhere.
//   D_MODEL=1024, N_HEADS=16, D_K=64, B=4, S=2048  -> M = B*S = 8192 rows.
// Pipeline:
//   1-3) Q/K/V = X @ W^T + b      (WMMA bf16 GEMM, fp32->bf16 on the fly,
//                                  128x64 blocks, software-pipelined loads)
//   4)   flash-style causal attention (WMMA scores + WMMA P*V, online
//        softmax; Q/K tiles staged via async global->LDS)
//   5)   out = O @ Wo^T + bo      (WMMA bf16 GEMM, fp32 output)
// ---------------------------------------------------------------------------

typedef __attribute__((ext_vector_type(16))) __bf16 v16bf;
typedef __attribute__((ext_vector_type(8)))  __bf16 bf16x8;
typedef __attribute__((ext_vector_type(8)))  float  v8f;
typedef __attribute__((ext_vector_type(4)))  float  f32x4;
typedef __attribute__((__vector_size__(16))) int    i32x4v;  // GCC-style, matches builtin proto

#define D_MODEL 1024
#define SEQ     2048
#define BATCH   4
#define NHEADS  16
#define DK      64
#define MROWS   (BATCH * SEQ)   // 8192

// ---------------- async global -> LDS (CDNA5), guarded -------------------
#if __has_builtin(__builtin_amdgcn_global_load_async_to_lds_b128) && \
    __has_builtin(__builtin_amdgcn_s_wait_asynccnt)
#define HAVE_ASYNC_LDS 1
#else
#define HAVE_ASYNC_LDS 0
#endif

__device__ __forceinline__ void cp16_g2l(const __bf16* g, __bf16* l) {
#if HAVE_ASYNC_LDS
  __builtin_amdgcn_global_load_async_to_lds_b128(
      (__attribute__((address_space(1))) i32x4v*)g,
      (__attribute__((address_space(3))) i32x4v*)l, 0, 0);
#else
  *(bf16x8*)l = *(const bf16x8*)g;
#endif
}
__device__ __forceinline__ void async_wait0() {
#if HAVE_ASYNC_LDS
  __builtin_amdgcn_s_wait_asynccnt(0);
#endif
}

// ---------------- WMMA helpers (layouts per CDNA5 ISA) ----------

__device__ __forceinline__ v8f wmma_bf16(v16bf a, v16bf b, v8f c) {
  // D = A(16x32 bf16) * B(32x16 bf16) + C(16x16 f32)
  return __builtin_amdgcn_wmma_f32_16x16x32_bf16(
      /*neg_a=*/false, a, /*neg_b=*/false, b,
      /*c_mod=*/(short)0, c, /*reuse_a=*/false, /*reuse_b=*/false);
}

// A fragment: 16x32 bf16 tile from row-major LDS (row stride ld elements).
// lane<16: row=lane,     elems 0..7 -> K=k0+0..7,  elems 8..15 -> K=k0+16..23
// lane>=16: row=lane-16, elems 0..7 -> K=k0+8..15, elems 8..15 -> K=k0+24..31
__device__ __forceinline__ v16bf load_a_frag(const __bf16* base, int ld,
                                             int lane, int k0) {
  const int row = lane & 15;
  const int kb  = k0 + ((lane & 16) ? 8 : 0);
  const bf16x8 lo = *(const bf16x8*)(base + row * ld + kb);
  const bf16x8 hi = *(const bf16x8*)(base + row * ld + kb + 16);
  v16bf a;
#pragma unroll
  for (int i = 0; i < 8; ++i) { a[i] = lo[i]; a[i + 8] = hi[i]; }
  return a;
}

// B fragment: 32x16 bf16 from a row-major [N x K] source (row n contiguous
// in K).  lane<16: n=lane, K=k0+0..7,16..23;  lane>=16: n=lane-16, K=k0+8..15,24..31.
__device__ __forceinline__ v16bf load_b_frag(const __bf16* base, int ld,
                                             int lane, int k0) {
  const int n  = lane & 15;
  const int kb = k0 + ((lane & 16) ? 8 : 0);
  const bf16x8 lo = *(const bf16x8*)(base + n * ld + kb);
  const bf16x8 hi = *(const bf16x8*)(base + n * ld + kb + 16);
  v16bf b;
#pragma unroll
  for (int i = 0; i < 8; ++i) { b[i] = lo[i]; b[i + 8] = hi[i]; }
  return b;
}

// ----- 8-element register staging (with fp32->bf16 convert if needed) -----
struct Pref8F { f32x4 lo, hi; };
struct Pref8B { bf16x8 v; };

__device__ __forceinline__ Pref8F load8g(const float* p) {
  Pref8F r; r.lo = *(const f32x4*)p; r.hi = *(const f32x4*)(p + 4); return r;
}
__device__ __forceinline__ Pref8B load8g(const __bf16* p) {
  Pref8B r; r.v = *(const bf16x8*)p; return r;
}
__device__ __forceinline__ void store8l(const Pref8F& r, __bf16* d) {
  bf16x8 v;
#pragma unroll
  for (int i = 0; i < 4; ++i) { v[i] = (__bf16)r.lo[i]; v[4 + i] = (__bf16)r.hi[i]; }
  *(bf16x8*)d = v;
}
__device__ __forceinline__ void store8l(const Pref8B& r, __bf16* d) {
  *(bf16x8*)d = r.v;
}

// ---------------- GEMM: Y[M,N] = X[M,K] @ W[N,K]^T + bias ------------------
// 128x64 block, 256 threads = 8 waves.  Wave w owns rows [w*16, w*16+16)
// and all 64 columns (4 accumulators); A fragment reused across 4 WMMAs.
// Global loads for the next k-slab are software-pipelined into registers.
#define BK  32
#define LDT 40  // 32 + 8 pad (bank spread, keeps 16B alignment)

template <typename XT, typename OT>
__global__ __launch_bounds__(256) void gemm_xwT(
    const XT* __restrict__ X, const float* __restrict__ W,
    const float* __restrict__ bias, OT* __restrict__ Y,
    int M, int N, int K) {
  __shared__ __attribute__((aligned(16))) __bf16 As[128][LDT];  // 10 KB
  __shared__ __attribute__((aligned(16))) __bf16 Bs[64][LDT];   //  5 KB

  const int tid  = threadIdx.x;
  const int lane = tid & 31;
  const int w    = tid >> 5;        // 0..7
  const int m0   = blockIdx.y * 128;
  const int n0   = blockIdx.x * 64;

  // staging geometry: A 128x32 (16 elts/thread), B 64x32 (8 elts/thread)
  const int arow = tid >> 1, ak = (tid & 1) * 16;  // ak in {0,16}
  const int brow = tid >> 2, bk = (tid & 3) * 8;   // bk in {0,8,16,24}

  const XT*    Ap = X + (size_t)(m0 + arow) * K;
  const float* Bp = W + (size_t)(n0 + brow) * K;

  auto ra0 = load8g(Ap + ak);
  auto ra1 = load8g(Ap + ak + 8);
  auto rb  = load8g(Bp + bk);

  v8f acc[4] = {v8f{}, v8f{}, v8f{}, v8f{}};

  for (int k0 = 0; k0 < K; k0 += BK) {
    store8l(ra0, &As[arow][ak]);
    store8l(ra1, &As[arow][ak + 8]);
    store8l(rb,  &Bs[brow][bk]);
    __syncthreads();

    const int kn = k0 + BK;
    if (kn < K) {  // prefetch next slab while WMMAs run
      ra0 = load8g(Ap + kn + ak);
      ra1 = load8g(Ap + kn + ak + 8);
      rb  = load8g(Bp + kn + bk);
    }

    const v16bf a = load_a_frag(&As[w * 16][0], LDT, lane, 0);
#pragma unroll
    for (int t = 0; t < 4; ++t) {
      const v16bf b = load_b_frag(&Bs[t * 16][0], LDT, lane, 0);
      acc[t] = wmma_bf16(a, b, acc[t]);
    }
    __syncthreads();
  }

  // Epilogue.  C layout: row = v + 8*(lane>=16), col = lane&15.
  const int col   = lane & 15;
  const int rbase = (lane >> 4) * 8;
#pragma unroll
  for (int t = 0; t < 4; ++t) {
    const float bb = bias[n0 + t * 16 + col];
#pragma unroll
    for (int v = 0; v < 8; ++v) {
      const size_t row = (size_t)(m0 + w * 16 + rbase + v);
      Y[row * N + n0 + t * 16 + col] = (OT)(acc[t][v] + bb);
    }
  }
}

// ---------------- Flash-style causal attention -----------------------------
// Block: 128 query rows of one (b,h).  8 waves; wave w owns q-rows
// [w*16, w*16+16) so softmax row stats are wave-local.
#define LDQ 72  // 64 + 8 pad

__global__ __launch_bounds__(256) void attn_causal(
    const __bf16* __restrict__ Qb, const __bf16* __restrict__ Kb,
    const __bf16* __restrict__ Vb, __bf16* __restrict__ Ob) {
  __shared__ __attribute__((aligned(16))) __bf16 Qs[128][LDQ];     // 18 KB
  __shared__ __attribute__((aligned(16))) __bf16 Ks[64][LDQ];      //  9 KB
  __shared__ __attribute__((aligned(16))) __bf16 Vt[64][LDQ];      //  9 KB (Vt[d][key])
  __shared__ __attribute__((aligned(16))) __bf16 Ps[8][16][LDQ];   // 18 KB (per-wave P)

  const int tid  = threadIdx.x;
  const int lane = tid & 31;
  const int w    = tid >> 5;
  const int bh   = blockIdx.y;
  const int b    = bh >> 4;
  const int h    = bh & 15;
  const int q0   = blockIdx.x * 128;

  // ---- stage Q (128 x 64 bf16), async -> LDS ----
  const __bf16* Qp = Qb + ((size_t)(b * SEQ + q0)) * D_MODEL + h * DK;
  {
    const int r = tid >> 1;
#pragma unroll
    for (int i = 0; i < 4; ++i) {
      const int d = (tid & 1) * 32 + i * 8;
      cp16_g2l(Qp + (size_t)r * D_MODEL + d, &Qs[r][d]);
    }
  }
  async_wait0();
  __syncthreads();

  // Resident Q fragments for this wave (k-steps 0 and 32 of d_k=64).
  const v16bf qa0 = load_a_frag(&Qs[w * 16][0], LDQ, lane, 0);
  const v16bf qa1 = load_a_frag(&Qs[w * 16][0], LDQ, lane, 32);

  v8f o[4] = {v8f{}, v8f{}, v8f{}, v8f{}};
  float m_[8], l_[8];
#pragma unroll
  for (int v = 0; v < 8; ++v) { m_[v] = -1e38f; l_[v] = 0.0f; }

  const int col   = lane & 15;
  const int rbase = (lane >> 4) * 8;
  const int nkb   = (q0 >> 6) + 2;  // causal: key blocks 0 .. (q0+127)/64

  for (int kb = 0; kb < nkb; ++kb) {
    const int kbase = kb * 64;
    const __bf16* Kp = Kb + ((size_t)(b * SEQ + kbase)) * D_MODEL + h * DK;
    const __bf16* Vp = Vb + ((size_t)(b * SEQ + kbase)) * D_MODEL + h * DK;

    __syncthreads();  // everyone done reading previous K/V tiles
    // stage K (64 x 64, row-major), async -> LDS
    {
      const int r = tid >> 2;
#pragma unroll
      for (int i = 0; i < 2; ++i) {
        const int d = (tid & 3) * 16 + i * 8;
        cp16_g2l(Kp + (size_t)r * D_MODEL + d, &Ks[r][d]);
      }
    }
    // stage V transposed: Vt[d][key]  (manual transpose, register path)
    {
      const int d    = tid & 63;
      const int keyb = (tid >> 6) * 16;
      bf16x8 v0, v1;
#pragma unroll
      for (int i = 0; i < 8; ++i) v0[i] = Vp[(size_t)(keyb + i) * D_MODEL + d];
#pragma unroll
      for (int i = 0; i < 8; ++i) v1[i] = Vp[(size_t)(keyb + 8 + i) * D_MODEL + d];
      *(bf16x8*)&Vt[d][keyb]     = v0;
      *(bf16x8*)&Vt[d][keyb + 8] = v1;
    }
    async_wait0();
    __syncthreads();

    // ---- scores S = Q * K^T  (16 q-rows x 64 keys per wave) ----
    v8f s[4];
#pragma unroll
    for (int t = 0; t < 4; ++t) {
      const v16bf bk0 = load_b_frag(&Ks[t * 16][0], LDQ, lane, 0);
      const v16bf bk1 = load_b_frag(&Ks[t * 16][0], LDQ, lane, 32);
      v8f acc = {};
      acc = wmma_bf16(qa0, bk0, acc);
      acc = wmma_bf16(qa1, bk1, acc);
      s[t] = acc;
    }

    // scale (+ causal mask only on the <=2 diagonal blocks of this wave)
    const bool needMask = (kbase + 63) > (q0 + w * 16);
    if (needMask) {
#pragma unroll
      for (int t = 0; t < 4; ++t) {
#pragma unroll
        for (int v = 0; v < 8; ++v) {
          const int gq = q0 + w * 16 + rbase + v;
          const int gk = kbase + t * 16 + col;
          const float val = s[t][v] * 0.125f;  // 1/sqrt(64)
          s[t][v] = (gk <= gq) ? val : -1e30f;
        }
      }
    } else {
#pragma unroll
      for (int t = 0; t < 4; ++t)
#pragma unroll
        for (int v = 0; v < 8; ++v) s[t][v] *= 0.125f;
    }

    // ---- online softmax update (row stats live in each 16-lane half) ----
    float alpha[8];
#pragma unroll
    for (int v = 0; v < 8; ++v) {
      float mx = fmaxf(fmaxf(s[0][v], s[1][v]), fmaxf(s[2][v], s[3][v]));
      mx = fmaxf(mx, __shfl_xor(mx, 1, 32));
      mx = fmaxf(mx, __shfl_xor(mx, 2, 32));
      mx = fmaxf(mx, __shfl_xor(mx, 4, 32));
      mx = fmaxf(mx, __shfl_xor(mx, 8, 32));
      const float mn = fmaxf(m_[v], mx);
      const float al = __expf(m_[v] - mn);
      float rs = 0.0f;
#pragma unroll
      for (int t = 0; t < 4; ++t) {
        const float p = __expf(s[t][v] - mn);
        s[t][v] = p;
        rs += p;
      }
      rs += __shfl_xor(rs, 1, 32);
      rs += __shfl_xor(rs, 2, 32);
      rs += __shfl_xor(rs, 4, 32);
      rs += __shfl_xor(rs, 8, 32);
      l_[v] = l_[v] * al + rs;
      m_[v] = mn;
      alpha[v] = al;
    }
#pragma unroll
    for (int t = 0; t < 4; ++t)
#pragma unroll
      for (int v = 0; v < 8; ++v) o[t][v] *= alpha[v];

    // ---- re-layout P (C-format) -> A-format via this wave's LDS patch ----
    // (LDS ops are in-order within a wave; no block barrier needed.)
#pragma unroll
    for (int t = 0; t < 4; ++t)
#pragma unroll
      for (int v = 0; v < 8; ++v)
        Ps[w][rbase + v][t * 16 + col] = (__bf16)s[t][v];

    const v16bf pa0 = load_a_frag(&Ps[w][0][0], LDQ, lane, 0);
    const v16bf pa1 = load_a_frag(&Ps[w][0][0], LDQ, lane, 32);

    // ---- O += P * V ----
#pragma unroll
    for (int t = 0; t < 4; ++t) {
      const v16bf bv0 = load_b_frag(&Vt[t * 16][0], LDQ, lane, 0);
      const v16bf bv1 = load_b_frag(&Vt[t * 16][0], LDQ, lane, 32);
      o[t] = wmma_bf16(pa0, bv0, o[t]);
      o[t] = wmma_bf16(pa1, bv1, o[t]);
    }
  }

  // ---- normalize + store (bf16, concat-head layout [b*S+q][h*64+d]) ----
  __bf16* Op = Ob + ((size_t)(b * SEQ + q0 + w * 16)) * D_MODEL + h * DK;
#pragma unroll
  for (int v = 0; v < 8; ++v) {
    const float inv = 1.0f / l_[v];
#pragma unroll
    for (int t = 0; t < 4; ++t)
      Op[(size_t)(rbase + v) * D_MODEL + t * 16 + col] =
          (__bf16)(o[t][v] * inv);
  }
}

// ---------------------------------------------------------------------------

extern "C" void kernel_launch(void* const* d_in, const int* in_sizes, int n_in,
                              void* d_out, int out_size, void* d_ws,
                              size_t ws_size, hipStream_t stream) {
  (void)in_sizes; (void)n_in; (void)out_size; (void)ws_size;
  const float* query  = (const float*)d_in[0];
  const float* key_in = (const float*)d_in[1];
  const float* value  = (const float*)d_in[2];
  const float* Wq = (const float*)d_in[3];
  const float* bq = (const float*)d_in[4];
  const float* Wk = (const float*)d_in[5];
  const float* bk = (const float*)d_in[6];
  const float* Wv = (const float*)d_in[7];
  const float* bv = (const float*)d_in[8];
  const float* Wo = (const float*)d_in[9];
  const float* bo = (const float*)d_in[10];
  float* out = (float*)d_out;

  const size_t NELT = (size_t)MROWS * D_MODEL;  // 8388608
  __bf16* Qb = (__bf16*)d_ws;
  __bf16* Kb = Qb + NELT;
  __bf16* Vb = Kb + NELT;
  __bf16* Ob = Vb + NELT;

  const dim3 gGemm(D_MODEL / 64, MROWS / 128);  // (16, 64)
  const dim3 gAttn(SEQ / 128, BATCH * NHEADS);  // (16, 64)

  gemm_xwT<float, __bf16><<<gGemm, 256, 0, stream>>>(query,  Wq, bq, Qb,
                                                     MROWS, D_MODEL, D_MODEL);
  gemm_xwT<float, __bf16><<<gGemm, 256, 0, stream>>>(key_in, Wk, bk, Kb,
                                                     MROWS, D_MODEL, D_MODEL);
  gemm_xwT<float, __bf16><<<gGemm, 256, 0, stream>>>(value,  Wv, bv, Vb,
                                                     MROWS, D_MODEL, D_MODEL);
  attn_causal<<<gAttn, 256, 0, stream>>>(Qb, Kb, Vb, Ob);
  gemm_xwT<__bf16, float><<<gGemm, 256, 0, stream>>>(Ob, Wo, bo, out,
                                                     MROWS, D_MODEL, D_MODEL);
}
